// ConvChannelToInstModel_15805479649346
// MI455X (gfx1250) — compile-verified, matrix operand order checked by reading, not run on hardware
//
#include <hip/hip_runtime.h>
#include <hip/hip_bf16.h>
#include <hip/hip_fp16.h>

// ---------------- problem constants ----------------
#define N_INST 128
#define N_NOTES 88
#define BATCH 64
#define TLEN 201
#define FLEN 52
#define T1 199   // after conv1
#define F1 50
#define T2 197   // after conv2
#define F2 48
#define FAFTER (T2*F2)        // 9456
#define NPAD 96               // 88 padded to 6 tiles of 16
#define KCHUNK 64             // halves per LDS chunk (2 WMMA k-steps)
#define NCHUNK 148            // ceil(9456/64)
#define KSPLIT 4              // 148 = 4 * 37
#define CHUNKS_PER_SEG 37
#define PITCH 72              // LDS row pitch in halves (144B, conflict-skewed)

typedef __attribute__((ext_vector_type(16))) _Float16 v16h;
typedef __attribute__((ext_vector_type(8)))  float    v8f;

// ---------------- workspace layout -----------------
// [0)                 xT  : f16 [I][B][FAFTER]          = 154,927,104 B
// [XT_BYTES)          acc : f32 [KSPLIT][I][B][NPAD]    =  12,582,912 B
#define XT_BYTES ((size_t)N_INST * BATCH * FAFTER * 2)

// =====================================================================
// Kernel 1: conv1 (1->I) + ReLU + depthwise conv2 + ReLU, f16 output
// grid = B*I blocks, 256 threads. y1 staged in LDS (39.8 KB).
// =====================================================================
__global__ __launch_bounds__(256) void conv_fused_kernel(
    const float* __restrict__ data, const float* __restrict__ w1,
    const float* __restrict__ b1,   const float* __restrict__ w2,
    const float* __restrict__ b2,   _Float16* __restrict__ xT) {
  const int i = blockIdx.x & (N_INST - 1);
  const int b = blockIdx.x >> 7;
  __shared__ float y1[T1 * F1];

  float W1[9], W2[9];
#pragma unroll
  for (int k = 0; k < 9; ++k) { W1[k] = w1[i * 9 + k]; W2[k] = w2[i * 9 + k]; }
  const float B1 = b1[i], B2 = b2[i];
  const float* __restrict__ db = data + (size_t)b * TLEN * FLEN;

  for (int idx = threadIdx.x; idx < T1 * F1; idx += 256) {
    const int t = idx / F1, f = idx % F1;
    float s = B1;
#pragma unroll
    for (int r = 0; r < 3; ++r)
#pragma unroll
      for (int c = 0; c < 3; ++c)
        s = __builtin_fmaf(db[(t + r) * FLEN + (f + c)], W1[r * 3 + c], s);
    y1[idx] = s > 0.f ? s : 0.f;
  }
  __syncthreads();

  _Float16* __restrict__ xo = xT + ((size_t)i * BATCH + b) * FAFTER;
  for (int idx = threadIdx.x; idx < T2 * F2; idx += 256) {
    const int t = idx / F2, f = idx % F2;
    float s = B2;
#pragma unroll
    for (int r = 0; r < 3; ++r)
#pragma unroll
      for (int c = 0; c < 3; ++c)
        s = __builtin_fmaf(y1[(t + r) * F1 + (f + c)], W2[r * 3 + c], s);
    s = s > 0.f ? s : 0.f;
    xo[idx] = (_Float16)s;
  }
}

// 16-byte-pair fragment gather matching the CDNA5 16-bit A/B VGPR layout:
// lane L holds row (L&15); lanes 0-15: K [0..7]+[16..23], lanes 16-31: +8.
__device__ __forceinline__ v16h load_frag16(const _Float16* __restrict__ base,
                                            int row, int kb) {
  const _Float16* q = base + row * PITCH + kb;
  union { v16h v; uint4 u[2]; } f;
  f.u[0] = *reinterpret_cast<const uint4*>(q);
  f.u[1] = *reinterpret_cast<const uint4*>(q + 16);
  return f.v;
}

// =====================================================================
// Kernel 2: per-instrument GEMM  out_seg[i] += x[:,i,:] * wfc[i]^T
// grid = (I, KSPLIT), 256 threads = 8 waves. Each wave owns 3 of the
// 4x6 16x16 tiles. wfc converted f32->f16 in registers, staged in LDS.
// =====================================================================
__global__ __launch_bounds__(256) void fc_wmma_kernel(
    const _Float16* __restrict__ xT, const float* __restrict__ wfc,
    float* __restrict__ acc) {
  const int i    = blockIdx.x;
  const int kseg = blockIdx.y;

  __shared__ _Float16 sm[(BATCH + NPAD) * PITCH];   // 23,040 B
  _Float16* sA = sm;                                 // 64 rows (M=batch)
  _Float16* sB = sm + BATCH * PITCH;                 // 96 rows (N=notes, padded)

  const int tid  = threadIdx.x;
  const int lane = tid & 31;
  const int w    = tid >> 5;       // 0..7
  const int mt   = w & 3;          // M tile 0..3
  const int ntb  = (w >> 2) * 3;   // N tile base: 0 or 3

  v8f c0 = {}, c1 = {}, c2 = {};

  const _Float16* __restrict__ xa = xT + (size_t)i * BATCH * FAFTER;
  const float*    __restrict__ wb = wfc + (size_t)i * N_NOTES * FAFTER;

  const int arow = tid >> 2;           // 0..63, 4 threads per A row
  const int acol = (tid & 3) * 16;     // 16-half granule
  const int frow = lane & 15;
  const int kb   = (lane >> 4) * 8;

  const int ch0 = kseg * CHUNKS_PER_SEG;
  for (int ch = ch0; ch < ch0 + CHUNKS_PER_SEG; ++ch) {
    const int k0 = ch * KCHUNK;

    // ---- stage A chunk (f16, 64x64) ----
    {
      uint4 va = {0, 0, 0, 0}, vb = {0, 0, 0, 0};
      const int k = k0 + acol;
      if (k < FAFTER) {   // 16 | FAFTER -> granule fully in or out
        const uint4* src =
            reinterpret_cast<const uint4*>(xa + (size_t)arow * FAFTER + k);
        va = src[0]; vb = src[1];
      }
      uint4* dst = reinterpret_cast<uint4*>(sA + arow * PITCH + acol);
      dst[0] = va; dst[1] = vb;
    }
    // ---- stage B chunk (f32 -> f16, 96x64; rows >=88 zero) ----
#pragma unroll
    for (int g = 0; g < 6; ++g) {
      const int gi  = tid + g * 256;     // 0..1535 granules of 4
      const int row = gi >> 4;
      const int col = (gi & 15) * 4;
      const int k   = k0 + col;
      float4 v = {0.f, 0.f, 0.f, 0.f};
      if (row < N_NOTES && k < FAFTER)
        v = *reinterpret_cast<const float4*>(wb + (size_t)row * FAFTER + k);
      union { _Float16 h[4]; uint2 u; } pk;
      pk.h[0] = (_Float16)v.x; pk.h[1] = (_Float16)v.y;
      pk.h[2] = (_Float16)v.z; pk.h[3] = (_Float16)v.w;
      *reinterpret_cast<uint2*>(sB + row * PITCH + col) = pk.u;
    }
    __syncthreads();

#pragma unroll
    for (int ks = 0; ks < KCHUNK; ks += 32) {
      v16h a  = load_frag16(sA, mt * 16 + frow,        ks + kb);
      v16h b0 = load_frag16(sB, (ntb + 0) * 16 + frow, ks + kb);
      v16h b1 = load_frag16(sB, (ntb + 1) * 16 + frow, ks + kb);
      v16h b2 = load_frag16(sB, (ntb + 2) * 16 + frow, ks + kb);
      c0 = __builtin_amdgcn_wmma_f32_16x16x32_f16(false, a, false, b0,
                                                  (short)0, c0, false, false);
      c1 = __builtin_amdgcn_wmma_f32_16x16x32_f16(false, a, false, b1,
                                                  (short)0, c1, false, false);
      c2 = __builtin_amdgcn_wmma_f32_16x16x32_f16(false, a, false, b2,
                                                  (short)0, c2, false, false);
    }
    __syncthreads();
  }

  // ---- write per-segment partials (plain stores: deterministic) ----
  // C layout: VGPR r, lanes 0-15 -> M=r, lanes 16-31 -> M=8+r; N=lane&15.
  float* __restrict__ as =
      acc + ((size_t)kseg * N_INST + i) * BATCH * NPAD;
  const int brow0 = mt * 16 + (lane >> 4) * 8;
  const int ncol  = lane & 15;
#pragma unroll
  for (int r = 0; r < 8; ++r) {
    const int bb = brow0 + r;
    as[(size_t)bb * NPAD + (ntb + 0) * 16 + ncol] = c0[r];
    as[(size_t)bb * NPAD + (ntb + 1) * 16 + ncol] = c1[r];
    as[(size_t)bb * NPAD + (ntb + 2) * 16 + ncol] = c2[r];
  }
}

// =====================================================================
// Kernel 3: reduce K-segments + bias + sigmoid -> out[b][i][n] f32
// =====================================================================
__global__ __launch_bounds__(256) void fc_epilogue_kernel(
    const float* __restrict__ acc, const float* __restrict__ bfc,
    float* __restrict__ out) {
  const int idx = blockIdx.x * 256 + threadIdx.x;   // over B*I*88
  if (idx >= BATCH * N_INST * N_NOTES) return;
  const int n = idx % N_NOTES;
  const int t = idx / N_NOTES;
  const int i = t % N_INST;
  const int b = t / N_INST;
  float v = bfc[i * N_NOTES + n];
#pragma unroll
  for (int s = 0; s < KSPLIT; ++s)
    v += acc[(((size_t)s * N_INST + i) * BATCH + b) * NPAD + n];
  out[idx] = 1.f / (1.f + __expf(-v));
}

// =====================================================================
extern "C" void kernel_launch(void* const* d_in, const int* in_sizes, int n_in,
                              void* d_out, int out_size, void* d_ws,
                              size_t ws_size, hipStream_t stream) {
  const float* data = (const float*)d_in[0];  // [64,1,201,52]
  const float* w1   = (const float*)d_in[1];  // [128,1,3,3]
  const float* b1   = (const float*)d_in[2];  // [128]
  const float* w2   = (const float*)d_in[3];  // [128,1,3,3]
  const float* b2   = (const float*)d_in[4];  // [128]
  const float* wfc  = (const float*)d_in[5];  // [128,88,9456]
  const float* bfc  = (const float*)d_in[6];  // [128,88]
  float* out = (float*)d_out;                 // [64,128,88]

  _Float16* xT = (_Float16*)d_ws;
  float* acc   = (float*)((char*)d_ws + XT_BYTES);

  conv_fused_kernel<<<BATCH * N_INST, 256, 0, stream>>>(data, w1, b1, w2, b2, xT);
  fc_wmma_kernel<<<dim3(N_INST, KSPLIT), 256, 0, stream>>>(xT, wfc, acc);
  const int nout = BATCH * N_INST * N_NOTES;
  fc_epilogue_kernel<<<(nout + 255) / 256, 256, 0, stream>>>(acc, bfc, out);
}
